// DiagonalSSMLayer_62302795596595
// MI455X (gfx1250) — matrix operand order repaired and checked by reading, not compiled, not run on hardware
//
#include <hip/hip_runtime.h>
#include <hip/hip_bf16.h>
#include <math.h>

// ---------------------------------------------------------------------------
// DiagonalSSMLayer on MI455X (gfx1250, wave32, WMMA bf16)
//   B=8, T=2048, d_model=n=1024, M = B*T = 16384, K = 1024 (fixed)
//   Pipeline:
//     cvt(f32->bf16: u, Wa, Wb, C)
//     gemm_gates : gates[m,n] = {sigmoid(u@Wa^T+ba), u@Wb^T+bb}  (fused, float2)
//     scan (3-pass chunked): hs = assoc-scan(gates) -> bf16
//     gemm_y     : y = hs@C^T + D.*u
// K-loops are unrolled x2 with two independent fragment buffers (no rotation
// copies -> no v_mov avalanche, no WMMA->VALU hazard NOPs in the hot loop).
// ---------------------------------------------------------------------------

typedef __attribute__((ext_vector_type(16))) __bf16 v16bf;
typedef __attribute__((ext_vector_type(8)))  __bf16 v8bf;
typedef __attribute__((ext_vector_type(8)))  float  v8f;

#define BATCH   8
#define TLEN    2048
#define DMODEL  1024
#define NDIM    1024
#define MROWS   (BATCH * TLEN)     // 16384
#define KDIM    1024
#define KITER2  (KDIM / 64)        // 16 double-iterations
#define CHUNKS  32
#define CLEN    (TLEN / CHUNKS)    // 64

// ---------------------------------------------------------------------------
// Fragment loaders (wave32; contiguous k order k=16*(l>>4)+s, same for A and B).
//  A (16x32, M x K): lane l -> row l&15;
//                    halves 0..15 at k+(l>>4)*16  (one 32B run)
//  B (32x16, K x N) from row-major weight W[N,K]: lane l -> W-row l&15;
//                    halves 0..15 at k+(l>>4)*16  (one 32B run)
// ---------------------------------------------------------------------------
__device__ __forceinline__ v16bf load_afrag(const __bf16* p) {
    v8bf lo = *(const v8bf*)(p);
    v8bf hi = *(const v8bf*)(p + 8);
    return __builtin_shufflevector(lo, hi,
        0, 1, 2, 3, 4, 5, 6, 7, 8, 9, 10, 11, 12, 13, 14, 15);
}
__device__ __forceinline__ v16bf load_bfrag(const __bf16* p) {
    v8bf lo = *(const v8bf*)(p);
    v8bf hi = *(const v8bf*)(p + 8);
    return __builtin_shufflevector(lo, hi,
        0, 1, 2, 3, 4, 5, 6, 7, 8, 9, 10, 11, 12, 13, 14, 15);
}
__device__ __forceinline__ v8f wmma_bf16(v16bf a, v16bf b, v8f c) {
    return __builtin_amdgcn_wmma_f32_16x16x32_bf16(false, a, false, b,
                                                   (short)0, c, false, false);
}

// ---------------------------------------------------------------------------
// f32 -> bf16 convert, 4 elements / thread
// ---------------------------------------------------------------------------
__global__ __launch_bounds__(256)
void cvt_f32_bf16_kernel(const float* __restrict__ in, __bf16* __restrict__ out,
                         size_t count4) {
    size_t i = (size_t)blockIdx.x * blockDim.x + threadIdx.x;
    if (i >= count4) return;
    const float4 v = ((const float4*)in)[i];
    ushort4 pk;
    pk.x = __builtin_bit_cast(unsigned short, (__bf16)v.x);
    pk.y = __builtin_bit_cast(unsigned short, (__bf16)v.y);
    pk.z = __builtin_bit_cast(unsigned short, (__bf16)v.z);
    pk.w = __builtin_bit_cast(unsigned short, (__bf16)v.w);
    ((ushort4*)out)[i] = pk;
}

// ---------------------------------------------------------------------------
// Fused gate GEMM: gates[m, n] = { sigmoid((u@Wa^T)[m,n] + ba[n]),
//                                  (u@Wb^T)[m,n] + bb[n] }   as float2
// Block 256 = 8 waves; block tile 128(M) x 64(N); wave tile 32x32 per weight.
// ---------------------------------------------------------------------------
__global__ __launch_bounds__(256)
void gemm_gates_kernel(const __bf16* __restrict__ A,   // M x K bf16
                       const __bf16* __restrict__ Wa,  // N x K bf16
                       const __bf16* __restrict__ Wb,  // N x K bf16
                       const float*  __restrict__ ba,  // [N]
                       const float*  __restrict__ bb,  // [N]
                       float2* __restrict__ gates) {   // M x N float2
    const int wave  = threadIdx.x >> 5;
    const int lane  = threadIdx.x & 31;
    const int hl    = lane >> 4;
    const int l16   = lane & 15;
    const int waveM = wave >> 1;                 // 0..3
    const int waveN = wave & 1;                  // 0..1
    const int m0 = blockIdx.x * 128 + waveM * 32;
    const int n0 = blockIdx.y * 64  + waveN * 32;

    const __bf16* pa[2];
    const __bf16* pwa[2];
    const __bf16* pwb[2];
    #pragma unroll
    for (int i = 0; i < 2; ++i) {
        pa[i]  = A  + (size_t)(m0 + i * 16 + l16) * KDIM + hl * 16;
        pwa[i] = Wa + (size_t)(n0 + i * 16 + l16) * KDIM + hl * 16;
        pwb[i] = Wb + (size_t)(n0 + i * 16 + l16) * KDIM + hl * 16;
    }

    const v8f vzero = {0.f, 0.f, 0.f, 0.f, 0.f, 0.f, 0.f, 0.f};
    v8f accA[2][2], accB[2][2];
    #pragma unroll
    for (int mi = 0; mi < 2; ++mi)
        #pragma unroll
        for (int ni = 0; ni < 2; ++ni) { accA[mi][ni] = vzero; accB[mi][ni] = vzero; }

    v16bf a0[2], wa0[2], wb0[2];   // ping
    v16bf a1[2], wa1[2], wb1[2];   // pong
    #pragma unroll
    for (int i = 0; i < 2; ++i) {
        a0[i]  = load_afrag(pa[i]);
        wa0[i] = load_bfrag(pwa[i]);
        wb0[i] = load_bfrag(pwb[i]);
    }

    #pragma unroll 1
    for (int kk = 0; kk < KITER2; ++kk) {
        const int kb = kk * 64;
        // prefetch k+32 into pong
        #pragma unroll
        for (int i = 0; i < 2; ++i) {
            a1[i]  = load_afrag(pa[i]  + kb + 32);
            wa1[i] = load_bfrag(pwa[i] + kb + 32);
            wb1[i] = load_bfrag(pwb[i] + kb + 32);
        }
        // 8 WMMAs on ping
        #pragma unroll
        for (int mi = 0; mi < 2; ++mi)
            #pragma unroll
            for (int ni = 0; ni < 2; ++ni) {
                accA[mi][ni] = wmma_bf16(a0[mi], wa0[ni], accA[mi][ni]);
                accB[mi][ni] = wmma_bf16(a0[mi], wb0[ni], accB[mi][ni]);
            }
        // prefetch k+64 into ping (uniform guard)
        if (kk + 1 < KITER2) {
            #pragma unroll
            for (int i = 0; i < 2; ++i) {
                a0[i]  = load_afrag(pa[i]  + kb + 64);
                wa0[i] = load_bfrag(pwa[i] + kb + 64);
                wb0[i] = load_bfrag(pwb[i] + kb + 64);
            }
        }
        // 8 WMMAs on pong
        #pragma unroll
        for (int mi = 0; mi < 2; ++mi)
            #pragma unroll
            for (int ni = 0; ni < 2; ++ni) {
                accA[mi][ni] = wmma_bf16(a1[mi], wa1[ni], accA[mi][ni]);
                accB[mi][ni] = wmma_bf16(a1[mi], wb1[ni], accB[mi][ni]);
            }
    }

    // epilogue: VGPR r -> row m0+mi*16+hl*8+r, col n0+ni*16+l16
    #pragma unroll
    for (int mi = 0; mi < 2; ++mi) {
        #pragma unroll
        for (int ni = 0; ni < 2; ++ni) {
            const int col = n0 + ni * 16 + l16;
            const float biasa = ba[col];
            const float biasb = bb[col];
            #pragma unroll
            for (int r = 0; r < 8; ++r) {
                const int row = m0 + mi * 16 + hl * 8 + r;
                float2 g;
                g.x = 1.0f / (1.0f + __expf(-(accA[mi][ni][r] + biasa)));
                g.y = accB[mi][ni][r] + biasb;
                gates[(size_t)row * NDIM + col] = g;   // b64 store, coalesced
            }
        }
    }
}

// ---------------------------------------------------------------------------
// Output GEMM: y[m,d] = (hs@C^T)[m,d] + D[d]*u[m,d]
// Block 256 = 8 waves; block tile 128(M) x 128(N); wave tile 32x64.
// ---------------------------------------------------------------------------
__global__ __launch_bounds__(256)
void gemm_y_kernel(const __bf16* __restrict__ A,   // M x K bf16 (hs)
                   const __bf16* __restrict__ W,   // N x K bf16 (C_w)
                   const float*  __restrict__ Dvec,// [N]
                   const float*  __restrict__ Uf,  // M x N f32
                   float* __restrict__ Out) {      // M x N f32
    const int wave  = threadIdx.x >> 5;
    const int lane  = threadIdx.x & 31;
    const int hl    = lane >> 4;
    const int l16   = lane & 15;
    const int waveM = wave >> 1;                 // 0..3
    const int waveN = wave & 1;                  // 0..1
    const int m0 = blockIdx.x * 128 + waveM * 32;
    const int n0 = blockIdx.y * 128 + waveN * 64;

    const __bf16* pa[2];
    const __bf16* pw[4];
    #pragma unroll
    for (int i = 0; i < 2; ++i)
        pa[i] = A + (size_t)(m0 + i * 16 + l16) * KDIM + hl * 16;
    #pragma unroll
    for (int i = 0; i < 4; ++i)
        pw[i] = W + (size_t)(n0 + i * 16 + l16) * KDIM + hl * 16;

    const v8f vzero = {0.f, 0.f, 0.f, 0.f, 0.f, 0.f, 0.f, 0.f};
    v8f acc[2][4];
    #pragma unroll
    for (int mi = 0; mi < 2; ++mi)
        #pragma unroll
        for (int ni = 0; ni < 4; ++ni) acc[mi][ni] = vzero;

    v16bf a0[2], b0[4];   // ping
    v16bf a1[2], b1[4];   // pong
    #pragma unroll
    for (int i = 0; i < 2; ++i) a0[i] = load_afrag(pa[i]);
    #pragma unroll
    for (int i = 0; i < 4; ++i) b0[i] = load_bfrag(pw[i]);

    #pragma unroll 1
    for (int kk = 0; kk < KITER2; ++kk) {
        const int kb = kk * 64;
        #pragma unroll
        for (int i = 0; i < 2; ++i) a1[i] = load_afrag(pa[i] + kb + 32);
        #pragma unroll
        for (int i = 0; i < 4; ++i) b1[i] = load_bfrag(pw[i] + kb + 32);
        #pragma unroll
        for (int mi = 0; mi < 2; ++mi)
            #pragma unroll
            for (int ni = 0; ni < 4; ++ni)
                acc[mi][ni] = wmma_bf16(a0[mi], b0[ni], acc[mi][ni]);
        if (kk + 1 < KITER2) {
            #pragma unroll
            for (int i = 0; i < 2; ++i) a0[i] = load_afrag(pa[i] + kb + 64);
            #pragma unroll
            for (int i = 0; i < 4; ++i) b0[i] = load_bfrag(pw[i] + kb + 64);
        }
        #pragma unroll
        for (int mi = 0; mi < 2; ++mi)
            #pragma unroll
            for (int ni = 0; ni < 4; ++ni)
                acc[mi][ni] = wmma_bf16(a1[mi], b1[ni], acc[mi][ni]);
    }

    #pragma unroll
    for (int mi = 0; mi < 2; ++mi) {
        #pragma unroll
        for (int ni = 0; ni < 4; ++ni) {
            const int col  = n0 + ni * 16 + l16;
            const float dm = Dvec[col];
            #pragma unroll
            for (int r = 0; r < 8; ++r) {
                const int row = m0 + mi * 16 + hl * 8 + r;
                Out[(size_t)row * DMODEL + col] =
                    acc[mi][ni][r] + dm * Uf[(size_t)row * DMODEL + col];
            }
        }
    }
}

// ---------------------------------------------------------------------------
// 3-pass chunked scan over T (h_t = a_t*h + b_t), composition:
//   (a2,b2) o (a1,b1) = (a2*a1, a2*b1 + b2)
// ---------------------------------------------------------------------------
__global__ __launch_bounds__(256)
void scan_chunk_reduce_kernel(const float2* __restrict__ gates,
                              float2* __restrict__ chunkAB) {  // [B,CHUNKS,n]
    const int tid = blockIdx.x * blockDim.x + threadIdx.x;
    const int ch = tid & (NDIM - 1);
    const int c  = (tid >> 10) & (CHUNKS - 1);
    const int b  = tid >> 15;
    const float2* g = gates + ((size_t)b * TLEN + c * CLEN) * NDIM + ch;
    float aprod = 1.f, bacc = 0.f;
    #pragma unroll 4
    for (int t = 0; t < CLEN; ++t) {
        const float2 ab = g[(size_t)t * NDIM];
        bacc  = ab.x * bacc + ab.y;
        aprod = aprod * ab.x;
    }
    chunkAB[tid] = make_float2(aprod, bacc);
}

__global__ __launch_bounds__(256)
void scan_carry_kernel(const float2* __restrict__ chunkAB,  // [B,CHUNKS,n]
                       float* __restrict__ carry) {         // [B,CHUNKS,n]
    const int tid = blockIdx.x * blockDim.x + threadIdx.x;  // B*n
    const int ch = tid & (NDIM - 1);
    const int b  = tid >> 10;
    float h = 0.f;
    #pragma unroll
    for (int c = 0; c < CHUNKS; ++c) {
        const size_t idx = ((size_t)b * CHUNKS + c) * NDIM + ch;
        carry[idx] = h;
        const float2 ab = chunkAB[idx];
        h = ab.x * h + ab.y;
    }
}

__global__ __launch_bounds__(256)
void scan_chunk_apply_kernel(const float2* __restrict__ gates,
                             const float* __restrict__ carry,
                             __bf16* __restrict__ hs) {     // [B,T,n] bf16
    const int tid = blockIdx.x * blockDim.x + threadIdx.x;
    const int ch = tid & (NDIM - 1);
    const int c  = (tid >> 10) & (CHUNKS - 1);
    const int b  = tid >> 15;
    const size_t base = ((size_t)b * TLEN + c * CLEN) * NDIM + ch;
    const float2* g = gates + base;
    float h = carry[tid];
    #pragma unroll 4
    for (int t = 0; t < CLEN; ++t) {
        const float2 ab = g[(size_t)t * NDIM];
        h = ab.x * h + ab.y;
        hs[base + (size_t)t * NDIM] = (__bf16)h;
    }
}

// ---------------------------------------------------------------------------
// Host-side orchestration
// ---------------------------------------------------------------------------
extern "C" void kernel_launch(void* const* d_in, const int* in_sizes, int n_in,
                              void* d_out, int out_size, void* d_ws, size_t ws_size,
                              hipStream_t stream) {
    (void)in_sizes; (void)n_in; (void)out_size; (void)ws_size;

    const float* u    = (const float*)d_in[0];  // [B,T,d]
    const float* Wa_w = (const float*)d_in[1];  // [n,d]
    const float* Wa_b = (const float*)d_in[2];  // [n]
    const float* Wb_w = (const float*)d_in[3];  // [n,d]
    const float* Wb_b = (const float*)d_in[4];  // [n]
    const float* C_w  = (const float*)d_in[5];  // [d,n]
    const float* Dv   = (const float*)d_in[6];  // [d]
    float* y = (float*)d_out;                   // [B,T,d]

    // Workspace layout (bytes from d_ws base)
    char* ws = (char*)d_ws;
    __bf16* u_bf    = (__bf16*)(ws);                    // 32 MB
    __bf16* hs_bf   = (__bf16*)(ws + 33554432ull);      // 32 MB
    __bf16* Wa_bf   = (__bf16*)(ws + 67108864ull);      //  2 MB
    __bf16* Wb_bf   = (__bf16*)(ws + 69206016ull);      //  2 MB
    __bf16* Cw_bf   = (__bf16*)(ws + 71303168ull);      //  2 MB
    float2* gates   = (float2*)(ws + 73400320ull);      // 128 MB
    float2* chunkAB = (float2*)(ws + 207618048ull);     //  2 MB
    float*  carry   = (float*) (ws + 209715200ull);     //  1 MB

    // 1) converts
    {
        const size_t cu = (size_t)MROWS * DMODEL / 4;
        cvt_f32_bf16_kernel<<<(unsigned)((cu + 255) / 256), 256, 0, stream>>>(u, u_bf, cu);
        const size_t cw = (size_t)NDIM * DMODEL / 4;
        cvt_f32_bf16_kernel<<<(unsigned)((cw + 255) / 256), 256, 0, stream>>>(Wa_w, Wa_bf, cw);
        cvt_f32_bf16_kernel<<<(unsigned)((cw + 255) / 256), 256, 0, stream>>>(Wb_w, Wb_bf, cw);
        cvt_f32_bf16_kernel<<<(unsigned)((cw + 255) / 256), 256, 0, stream>>>(C_w, Cw_bf, cw);
    }

    // 2) fused gate GEMM -> gates (float2 {alpha, Bu})
    gemm_gates_kernel<<<dim3(MROWS / 128, NDIM / 64), 256, 0, stream>>>(
        u_bf, Wa_bf, Wb_bf, Wa_b, Wb_b, gates);

    // 3) chunked scan -> hs (bf16)
    scan_chunk_reduce_kernel<<<(BATCH * CHUNKS * NDIM) / 256, 256, 0, stream>>>(gates, chunkAB);
    scan_carry_kernel<<<(BATCH * NDIM) / 256, 256, 0, stream>>>(chunkAB, carry);
    scan_chunk_apply_kernel<<<(BATCH * CHUNKS * NDIM) / 256, 256, 0, stream>>>(gates, carry, hs_bf);

    // 4) y = hs@C^T + D.*u
    gemm_y_kernel<<<dim3(MROWS / 128, DMODEL / 128), 256, 0, stream>>>(
        hs_bf, Cw_bf, Dv, u, y);
}
